// VGP_3728031613392
// MI455X (gfx1250) — compile-verified, matrix operand order checked by reading, not run on hardware
//
#include <hip/hip_runtime.h>
#include <hip/hip_bf16.h>
#include <math.h>

// ---------------------------------------------------------------- types
typedef __bf16 bf16_t;
typedef bf16_t v16bf __attribute__((ext_vector_type(16)));
typedef bf16_t v8bf  __attribute__((ext_vector_type(8)));
typedef bf16_t v4bf  __attribute__((ext_vector_type(4)));
typedef float  v8f   __attribute__((ext_vector_type(8)));
typedef float  v2f   __attribute__((ext_vector_type(2)));
typedef int    v4i   __attribute__((ext_vector_type(4)));

#define B_DIM 4096
#define X_DIM 1024
#define H_DIM 2048
#define T_DIM 64
#define D_DIM 16
#define Z_DIM 1024
#define NTOT  4096
#define LOG2PI_F 1.8378770664093453f

#if defined(__HIP_DEVICE_COMPILE__) && defined(__has_builtin)
#if __has_builtin(__builtin_amdgcn_global_load_async_to_lds_b128)
#define ASYNC_LDS 1
#endif
#endif

static inline int cdiv(int a, int b) { return (a + b - 1) / b; }

// issue one 64x32 bf16 tile (global -> LDS), 128 threads, 2 async b128/wave
__device__ __forceinline__ void issue_tile(const bf16_t* __restrict__ base, int ld,
                                           int row0, int k0, bf16_t (*dst)[40], int tid)
{
  #pragma unroll
  for (int i = tid; i < 256; i += 128) {
    int r = i >> 2, c8 = (i & 3) << 3;
    const bf16_t* g = base + (size_t)(row0 + r) * ld + (k0 + c8);
#ifdef ASYNC_LDS
    __builtin_amdgcn_global_load_async_to_lds_b128((v4i*)g, (v4i*)&dst[r][c8], 0, 0);
#else
    *(v8bf*)&dst[r][c8] = *(const v8bf*)g;
#endif
  }
}

// ================================================================
// bf16 WMMA GEMM:  C[M,N] = act(A[M,K] @ W[N,K]^T + bias)
// A, W bf16; optional f32 and/or bf16 outputs.
// 64x64 tile, 128 threads (4 waves), each wave 32x32 (2x2 WMMA frags).
// Interior tiles: double-buffered async global->LDS pipeline.
// ================================================================
__launch_bounds__(128)
__global__ void k_gemm_bf16(const bf16_t* __restrict__ A, int lda,
                            const bf16_t* __restrict__ W, int ldw,
                            const float* __restrict__ bias,
                            float* __restrict__ Cf, bf16_t* __restrict__ Cb, int ldc,
                            int M, int N, int K, int relu)
{
  __shared__ __align__(16) bf16_t As[2][64][40];   // 80B row stride: b128-aligned
  __shared__ __align__(16) bf16_t Bs[2][64][40];
  const int tid  = threadIdx.x;
  const int wave = tid >> 5, lane = tid & 31;
  const int lrow = lane & 15, lhalf = lane >> 4;
  const int m0g = blockIdx.x * 64, n0g = blockIdx.y * 64;
  const int wm = (wave & 1) * 32, wn = (wave >> 1) * 32;

  v8f acc[2][2] = {};

  const bool pipeOK = (m0g + 64 <= M) && (n0g + 64 <= N) && ((K & 31) == 0);

  if (pipeOK) {
    // ---------------- double-buffered async pipeline ----------------
    const int nch = K >> 5;
    issue_tile(A, lda, m0g, 0, As[0], tid);
    issue_tile(W, ldw, n0g, 0, Bs[0], tid);
    for (int ch = 0; ch < nch; ++ch) {
      const int cur = ch & 1;
      if (ch + 1 < nch) {
        issue_tile(A, lda, m0g, (ch + 1) << 5, As[cur ^ 1], tid);
        issue_tile(W, ldw, n0g, (ch + 1) << 5, Bs[cur ^ 1], tid);
#ifdef ASYNC_LDS
        asm volatile("s_wait_asynccnt 4" ::: "memory");   // chunk ch landed
#endif
      } else {
#ifdef ASYNC_LDS
        asm volatile("s_wait_asynccnt 0" ::: "memory");
#endif
      }
      __syncthreads();                       // all waves' chunk-ch data visible

      v16bf afr[2], bfr[2];
      #pragma unroll
      for (int i = 0; i < 2; ++i) {
        int ar = wm + i * 16 + lrow;
        int br = wn + i * 16 + lrow;
        #pragma unroll
        for (int e = 0; e < 8; ++e) {
          int klo = lhalf * 8 + e;
          afr[i][e]     = As[cur][ar][klo];
          afr[i][e + 8] = As[cur][ar][16 + klo];
          bfr[i][e]     = Bs[cur][br][klo];
          bfr[i][e + 8] = Bs[cur][br][16 + klo];
        }
      }
      #pragma unroll
      for (int mi = 0; mi < 2; ++mi)
        #pragma unroll
        for (int ni = 0; ni < 2; ++ni)
          acc[mi][ni] = __builtin_amdgcn_wmma_f32_16x16x32_bf16(
              false, afr[mi], false, bfr[ni], (short)0, acc[mi][ni], false, false);

      __syncthreads();                       // safe to overwrite buf[cur] later
    }
  } else {
    // ---------------- masked single-buffer path (edge tiles) ----------------
    for (int k0 = 0; k0 < K; k0 += 32) {
      __syncthreads();
      const bool inK = (k0 + 32 <= K);
      if ((m0g + 64 <= M) && inK) {
        issue_tile(A, lda, m0g, k0, As[0], tid);
      } else {
        for (int i = tid; i < 2048; i += 128) {
          int r = i >> 5, c = i & 31;
          int gr = m0g + r, gc = k0 + c;
          As[0][r][c] = (gr < M && gc < K) ? A[(size_t)gr * lda + gc] : (bf16_t)0.f;
        }
      }
      if ((n0g + 64 <= N) && inK) {
        issue_tile(W, ldw, n0g, k0, Bs[0], tid);
      } else {
        for (int i = tid; i < 2048; i += 128) {
          int r = i >> 5, c = i & 31;
          int gr = n0g + r, gc = k0 + c;
          Bs[0][r][c] = (gr < N && gc < K) ? W[(size_t)gr * ldw + gc] : (bf16_t)0.f;
        }
      }
#ifdef ASYNC_LDS
      asm volatile("s_wait_asynccnt 0" ::: "memory");
#endif
      __syncthreads();

      v16bf afr[2], bfr[2];
      #pragma unroll
      for (int i = 0; i < 2; ++i) {
        int ar = wm + i * 16 + lrow;
        int br = wn + i * 16 + lrow;
        #pragma unroll
        for (int e = 0; e < 8; ++e) {
          int klo = lhalf * 8 + e;
          afr[i][e]     = As[0][ar][klo];
          afr[i][e + 8] = As[0][ar][16 + klo];
          bfr[i][e]     = Bs[0][br][klo];
          bfr[i][e + 8] = Bs[0][br][16 + klo];
        }
      }
      #pragma unroll
      for (int mi = 0; mi < 2; ++mi)
        #pragma unroll
        for (int ni = 0; ni < 2; ++ni)
          acc[mi][ni] = __builtin_amdgcn_wmma_f32_16x16x32_bf16(
              false, afr[mi], false, bfr[ni], (short)0, acc[mi][ni], false, false);
    }
  }

  #pragma unroll
  for (int mi = 0; mi < 2; ++mi)
    #pragma unroll
    for (int ni = 0; ni < 2; ++ni)
      #pragma unroll
      for (int r = 0; r < 8; ++r) {
        int ml  = r + (lhalf ? 8 : 0);
        int row = m0g + wm + mi * 16 + ml;
        int col = n0g + wn + ni * 16 + lrow;
        if (row < M && col < N) {
          float v = acc[mi][ni][r];
          if (bias) v += bias[col];
          if (relu) v = v > 0.f ? v : 0.f;
          size_t idx = (size_t)row * ldc + col;
          if (Cf) Cf[idx] = v;
          if (Cb) Cb[idx] = (bf16_t)v;
        }
      }
}

// ================================================================
// fp32 WMMA GEMM (16x16x4_f32), generic operand transposes + epilogues
//   Aop[m,k] = aT  ? A[k*lda+m]  : A[m*lda+k]
//   Bop[k,n] = bKN ? B[k*ldb+n]  : B[n*ldb+k]
// mode 0: C = Aop*Bop
// mode 1: C = C - Aop*Bop
// mode 2: C = exp(-0.5*(v1[m]+v2[n]-2*Aop*Bop))       (RBF kernel)
// mode 3: C = sqrt(max(v1[m],0))*v2[m*ldE+n] + Aop*Bop (f = sqrt(cd)*eps + mu)
// ================================================================
__launch_bounds__(128)
__global__ void k_gemm_f32(const float* __restrict__ A, int lda, int aT,
                           const float* __restrict__ Bm, int ldb, int bKN,
                           float* __restrict__ C, int ldc,
                           int M, int N, int K, int mode, int lowerOnly,
                           const float* __restrict__ v1,
                           const float* __restrict__ v2, int ldE)
{
  const int m0g = blockIdx.x * 64, n0g = blockIdx.y * 64;
  if (lowerOnly && n0g > m0g) return;

  __shared__ __align__(16) float As[64][36];    // 144B row stride: f4-aligned
  __shared__ __align__(16) float Bs[64][36];
  const int tid  = threadIdx.x;
  const int wave = tid >> 5, lane = tid & 31;
  const int lrow = lane & 15, lhalf = lane >> 4;
  const int wm = (wave & 1) * 32, wn = (wave >> 1) * 32;
  const int koff = lhalf * 2;

  v8f acc[2][2] = {};

  for (int k0 = 0; k0 < K; k0 += 32) {
    __syncthreads();
    const bool inK = (k0 + 32 <= K);
    const bool inM = (m0g + 64 <= M);
    const bool inN = (n0g + 64 <= N);
    // ---- A tile
    if (!aT && inM && inK) {
      #pragma unroll
      for (int i = tid; i < 512; i += 128) {
        int r = i >> 3, c4 = (i & 7) << 2;
        *(float4*)&As[r][c4] = *(const float4*)(A + (size_t)(m0g + r) * lda + k0 + c4);
      }
    } else if (aT && inM && inK) {
      for (int i = tid; i < 2048; i += 128) {      // uniform, strided, no masks
        int r = i >> 5, c = i & 31;
        As[r][c] = A[(size_t)(k0 + c) * lda + (m0g + r)];
      }
    } else {
      for (int i = tid; i < 2048; i += 128) {
        int r = i >> 5, c = i & 31;
        int gm = m0g + r, gk = k0 + c;
        float v = 0.f;
        if (gm < M && gk < K)
          v = aT ? A[(size_t)gk * lda + gm] : A[(size_t)gm * lda + gk];
        As[r][c] = v;
      }
    }
    // ---- B tile (LDS layout: Bs[n][k])
    if (!bKN && inN && inK) {
      #pragma unroll
      for (int i = tid; i < 512; i += 128) {
        int r = i >> 3, c4 = (i & 7) << 2;
        *(float4*)&Bs[r][c4] = *(const float4*)(Bm + (size_t)(n0g + r) * ldb + k0 + c4);
      }
    } else if (bKN && inN && inK) {
      #pragma unroll
      for (int i = tid; i < 512; i += 128) {
        int kk = i >> 4, n4 = (i & 15) << 2;
        float4 v = *(const float4*)(Bm + (size_t)(k0 + kk) * ldb + (n0g + n4));
        Bs[n4 + 0][kk] = v.x; Bs[n4 + 1][kk] = v.y;
        Bs[n4 + 2][kk] = v.z; Bs[n4 + 3][kk] = v.w;
      }
    } else {
      for (int i = tid; i < 2048; i += 128) {
        int r = i >> 5, c = i & 31;
        int gn = n0g + r, gk = k0 + c;
        float v = 0.f;
        if (gn < N && gk < K)
          v = bKN ? Bm[(size_t)gk * ldb + gn] : Bm[(size_t)gn * ldb + gk];
        Bs[r][c] = v;
      }
    }
    __syncthreads();

    #pragma unroll
    for (int kk = 0; kk < 32; kk += 4) {
      v2f a2[2], b2[2];
      #pragma unroll
      for (int i = 0; i < 2; ++i) {
        int ar = wm + i * 16 + lrow;
        int br = wn + i * 16 + lrow;
        a2[i][0] = As[ar][kk + koff];
        a2[i][1] = As[ar][kk + koff + 1];
        b2[i][0] = Bs[br][kk + koff];
        b2[i][1] = Bs[br][kk + koff + 1];
      }
      #pragma unroll
      for (int mi = 0; mi < 2; ++mi)
        #pragma unroll
        for (int ni = 0; ni < 2; ++ni)
          acc[mi][ni] = __builtin_amdgcn_wmma_f32_16x16x4_f32(
              false, a2[mi], false, b2[ni], (short)0, acc[mi][ni], false, false);
    }
  }

  #pragma unroll
  for (int mi = 0; mi < 2; ++mi)
    #pragma unroll
    for (int ni = 0; ni < 2; ++ni)
      #pragma unroll
      for (int r = 0; r < 8; ++r) {
        int ml  = r + (lhalf ? 8 : 0);
        int row = m0g + wm + mi * 16 + ml;
        int col = n0g + wn + ni * 16 + lrow;
        if (row < M && col < N) {
          float v = acc[mi][ni][r];
          size_t idx = (size_t)row * ldc + col;
          if (mode == 0)      C[idx] = v;
          else if (mode == 1) C[idx] = C[idx] - v;
          else if (mode == 2) C[idx] = expf(-0.5f * (v1[row] + v2[col] - 2.f * v));
          else {
            float cd = v1[row]; cd = cd > 0.f ? cd : 0.f;
            C[idx] = sqrtf(cd) * v2[(size_t)row * ldE + col] + v;
          }
        }
      }
}

// ================================================================
// Cholesky 64x64 diagonal block (in place, lower), 64 threads
// ================================================================
__launch_bounds__(64)
__global__ void k_potf2(float* __restrict__ Kss, int ld, int jb)
{
  __shared__ float a[64][65];
  const int tid = threadIdx.x;
  for (int j = 0; j < 64; ++j) a[tid][j] = Kss[(size_t)(jb + tid) * ld + jb + j];
  __syncthreads();
  for (int j = 0; j < 64; ++j) {
    float d = sqrtf(a[j][j]);
    __syncthreads();
    if (tid == j)     a[j][j]   = d;
    else if (tid > j) a[tid][j] = a[tid][j] / d;
    __syncthreads();
    if (tid > j) {
      float lij = a[tid][j];
      for (int c = j + 1; c <= tid; ++c) a[tid][c] -= lij * a[c][j];
    }
    __syncthreads();
  }
  for (int j = 0; j < 64; ++j) Kss[(size_t)(jb + tid) * ld + jb + j] = a[tid][j];
}

// panel TRSM: rows below diag block:  Lp = A_panel * Ld^{-T}
__launch_bounds__(128)
__global__ void k_panel_trsm(float* __restrict__ Kss, int ld, int jb,
                             int rows0, int nrows)
{
  __shared__ float Ld[64][65];
  for (int i = threadIdx.x; i < 64 * 64; i += 128)
    Ld[i >> 6][i & 63] = Kss[(size_t)(jb + (i >> 6)) * ld + jb + (i & 63)];
  __syncthreads();
  int i = blockIdx.x * 128 + threadIdx.x;
  if (i >= nrows) return;
  size_t rb = (size_t)(rows0 + i) * ld + jb;
  float r[64];
  for (int j = 0; j < 64; ++j) r[j] = Kss[rb + j];
  for (int j = 0; j < 64; ++j) {
    float v = r[j];
    for (int t = 0; t < j; ++t) v -= r[t] * Ld[j][t];
    r[j] = v / Ld[j][j];
  }
  for (int j = 0; j < 64; ++j) Kss[rb + j] = r[j];
}

// block triangular solve on 64-row slab of Y (one thread per RHS column)
__launch_bounds__(128)
__global__ void k_trsm_diag(const float* __restrict__ Kss, int ld, int jb,
                            float* __restrict__ Y, int ldy, int ncols, int backward)
{
  __shared__ float Ld[64][65];
  for (int i = threadIdx.x; i < 64 * 64; i += 128)
    Ld[i >> 6][i & 63] = Kss[(size_t)(jb + (i >> 6)) * ld + jb + (i & 63)];
  __syncthreads();
  int c = blockIdx.x * 128 + threadIdx.x;
  if (c >= ncols) return;
  float y[64];
  for (int r = 0; r < 64; ++r) y[r] = Y[(size_t)(jb + r) * ldy + c];
  if (!backward) {
    for (int r = 0; r < 64; ++r) {
      float v = y[r];
      for (int t = 0; t < r; ++t) v -= Ld[r][t] * y[t];
      y[r] = v / Ld[r][r];
    }
  } else {
    for (int r = 63; r >= 0; --r) {
      float v = y[r];
      for (int t = r + 1; t < 64; ++t) v -= Ld[t][r] * y[t];
      y[r] = v / Ld[r][r];
    }
  }
  for (int r = 0; r < 64; ++r) Y[(size_t)(jb + r) * ldy + c] = y[r];
}

// ---------------------------------------------------------------- small utils
__global__ void k_cvt(const float* __restrict__ a, bf16_t* __restrict__ o, size_t n)
{
  size_t n4 = n >> 2;
  for (size_t i = (size_t)blockIdx.x * blockDim.x + threadIdx.x; i < n4;
       i += (size_t)gridDim.x * blockDim.x) {
    float4 v = ((const float4*)a)[i];
    v4bf b; b[0] = (bf16_t)v.x; b[1] = (bf16_t)v.y; b[2] = (bf16_t)v.z; b[3] = (bf16_t)v.w;
    ((v4bf*)o)[i] = b;
  }
}

__global__ void k_rownorm(const float* __restrict__ X, int ldx, int cols,
                          float* __restrict__ out, int nrows)
{
  int i = blockIdx.x * blockDim.x + threadIdx.x;
  if (i >= nrows) return;
  const float* p = X + (size_t)i * ldx;
  float s = 0.f;
  for (int j = 0; j < cols; ++j) s += p[j] * p[j];
  out[i] = s;
}

__global__ void k_copy_diag(const float* __restrict__ Mtx, int ld,
                            float* __restrict__ d, int n)
{
  int i = blockIdx.x * blockDim.x + threadIdx.x;
  if (i < n) d[i] = Mtx[(size_t)i * ld + i];
}

__global__ void k_cd(const float* __restrict__ Y, int ld,
                     const float* __restrict__ dK, float* __restrict__ cd, int n)
{
  int i = blockIdx.x * blockDim.x + threadIdx.x;
  if (i < n) cd[i] = 1.f - Y[(size_t)i * ld + i] * dK[i];
}

__global__ void k_concat_bf(const bf16_t* __restrict__ A, int lda, int ca,
                            const bf16_t* __restrict__ Bb, int ldb, int cb,
                            bf16_t* __restrict__ O, int ldo, int rows)
{
  size_t tot = (size_t)rows * (ca + cb);
  for (size_t idx = (size_t)blockIdx.x * blockDim.x + threadIdx.x; idx < tot;
       idx += (size_t)gridDim.x * blockDim.x) {
    size_t r = idx / (ca + cb);
    int c = (int)(idx % (ca + cb));
    O[r * ldo + c] = (c < ca) ? A[r * lda + c] : Bb[r * ldb + (c - ca)];
  }
}

__global__ void k_reparam(const float* __restrict__ zm, const float* __restrict__ zc,
                          const float* __restrict__ eps, bf16_t* __restrict__ z, size_t n)
{
  for (size_t i = (size_t)blockIdx.x * blockDim.x + threadIdx.x; i < n;
       i += (size_t)gridDim.x * blockDim.x)
    z[i] = (bf16_t)(expf(0.5f * zc[i]) * eps[i] + zm[i]);
}

// ---------------------------------------------------------------- reductions
#define RGRID 1024
__device__ __forceinline__ void block_red(float s, float* partial)
{
  __shared__ float sh[256];
  sh[threadIdx.x] = s; __syncthreads();
  for (int o = 128; o > 0; o >>= 1) {
    if (threadIdx.x < o) sh[threadIdx.x] += sh[threadIdx.x + o];
    __syncthreads();
  }
  if (threadIdx.x == 0) partial[blockIdx.x] = sh[0];
}

__global__ void k_red_kld(const float* __restrict__ mp, const float* __restrict__ lcp,
                          int ld, int col0, int cols, int rows, float* __restrict__ partial)
{
  size_t tot = (size_t)rows * cols;
  float s = 0.f;
  for (size_t idx = (size_t)blockIdx.x * blockDim.x + threadIdx.x; idx < tot;
       idx += (size_t)gridDim.x * blockDim.x) {
    size_t r = idx / cols; int c = (int)(idx % cols);
    size_t o = r * ld + col0 + c;
    float m = mp[o], lc = lcp[o];
    s += -lc - 1.f + expf(lc) + m * m;
  }
  block_red(s, partial);
}

__global__ void k_red_sq(const float* __restrict__ a, int lda_, int ca0,
                         const float* __restrict__ b, int ldb_, int cb0,
                         int cols, int rows, float* __restrict__ partial)
{
  size_t tot = (size_t)rows * cols;
  float s = 0.f;
  for (size_t idx = (size_t)blockIdx.x * blockDim.x + threadIdx.x; idx < tot;
       idx += (size_t)gridDim.x * blockDim.x) {
    size_t r = idx / cols; int c = (int)(idx % cols);
    float d = a[r * lda_ + ca0 + c] - b[r * ldb_ + cb0 + c];
    s += d * d;
  }
  block_red(s, partial);
}

__global__ void k_red_sumsq(const float* __restrict__ a, size_t n, float* __restrict__ partial)
{
  float s = 0.f;
  for (size_t i = (size_t)blockIdx.x * blockDim.x + threadIdx.x; i < n;
       i += (size_t)gridDim.x * blockDim.x) s += a[i] * a[i];
  block_red(s, partial);
}

__global__ void k_finalize(const float* __restrict__ P, float* __restrict__ out)
{
  if (threadIdx.x != 0 || blockIdx.x != 0) return;
  double kz = 0, kf = 0, x2 = 0, r2 = 0, q = 0;
  for (int i = 0; i < RGRID; ++i) kz += P[i];
  for (int i = 0; i < RGRID; ++i) kf += P[RGRID + i];
  for (int i = 0; i < RGRID; ++i) x2 += P[2 * RGRID + i];
  for (int i = 0; i < RGRID; ++i) r2 += P[3 * RGRID + i];
  for (int i = 0; i < RGRID; ++i) q  += P[4 * RGRID + i];
  double Bd = (double)B_DIM;
  double kld = 0.5 * (kz + kf) / Bd;
  double logq = -0.5 * q - 0.5 * (double)LOG2PI_F * Bd * (double)T_DIM;
  double nll = x2 / (Bd * (double)X_DIM) / Bd + logq - r2 / (Bd * (double)T_DIM) / Bd;
  out[0] = (float)kld;
  out[1] = (float)nll;
}

// ================================================================
// host-side orchestration
// ================================================================
extern "C" void kernel_launch(void* const* d_in, const int* in_sizes, int n_in,
                              void* d_out, int out_size, void* d_ws, size_t ws_size,
                              hipStream_t stream)
{
  (void)in_sizes; (void)n_in; (void)out_size; (void)ws_size;
  const float* x      = (const float*)d_in[0];
  const float* xi     = (const float*)d_in[1];
  const float* eps_gp = (const float*)d_in[2];
  const float* eps_nm = (const float*)d_in[3];
  const float* fw[12] = { (const float*)d_in[4],  (const float*)d_in[6],
                          (const float*)d_in[8],  (const float*)d_in[10],
                          (const float*)d_in[12], (const float*)d_in[14],
                          (const float*)d_in[16], (const float*)d_in[18],
                          (const float*)d_in[20], (const float*)d_in[22],
                          (const float*)d_in[24], (const float*)d_in[26] };
  const float* fb[12] = { (const float*)d_in[5],  (const float*)d_in[7],
                          (const float*)d_in[9],  (const float*)d_in[11],
                          (const float*)d_in[13], (const float*)d_in[15],
                          (const float*)d_in[17], (const float*)d_in[19],
                          (const float*)d_in[21], (const float*)d_in[23],
                          (const float*)d_in[25], (const float*)d_in[27] };
  const size_t wsz[12] = { (size_t)H_DIM*X_DIM, (size_t)T_DIM*H_DIM, (size_t)D_DIM*H_DIM,
                           (size_t)H_DIM*80,    (size_t)Z_DIM*H_DIM, (size_t)Z_DIM*H_DIM,
                           (size_t)H_DIM*2048,  (size_t)80*H_DIM,    (size_t)80*H_DIM,
                           (size_t)H_DIM*Z_DIM, (size_t)X_DIM*H_DIM, (size_t)X_DIM*H_DIM };

  // ---- workspace layout
  float* ws = (float*)d_ws;
  float* Kss  = ws;                   // 4096*4096
  float* Y    = Kss  + 16777216;      // 4096*4096
  float* sbuf = Y    + 16777216;      // 4096*64
  float* tbuf = sbuf + 262144;        // 4096*16
  float* fbuf = tbuf + 65536;         // 4096*16
  float* encm = fbuf + 65536;         // 4096*80
  float* encc = encm + 327680;        // 4096*80
  float* sn   = encc + 327680;        // 4096
  float* xin  = sn   + 4096;
  float* dK   = xin  + 4096;
  float* cd   = dK   + 4096;
  float* part = cd   + 4096;          // 5*RGRID
  bf16_t* bp  = (bf16_t*)(part + 5 * RGRID + 64);
  bf16_t* x_bf      = bp;             bp += (size_t)B_DIM * X_DIM;
  bf16_t* xi_bf     = bp;             bp += (size_t)B_DIM * T_DIM;
  bf16_t* h1_bf     = bp;             bp += (size_t)B_DIM * H_DIM;
  bf16_t* hbuf_bf   = bp;             bp += (size_t)B_DIM * H_DIM;   // h2/h23/h3
  bf16_t* z_bf      = bp;             bp += (size_t)B_DIM * Z_DIM;
  bf16_t* catxz_bf  = bp;             bp += (size_t)B_DIM * 2048;
  bf16_t* catfxi_bf = bp;             bp += (size_t)B_DIM * 80;
  bf16_t* f_bf      = bp;             bp += (size_t)B_DIM * D_DIM;
  bf16_t* wbf[12];
  for (int i = 0; i < 12; ++i) { wbf[i] = bp; bp += wsz[i]; }

  float* out  = (float*)d_out;
  float* zm_o = out + 2;
  float* zc_o = zm_o + (size_t)B_DIM * Z_DIM;
  float* xm_o = zc_o + (size_t)B_DIM * Z_DIM;
  float* xc_o = xm_o + (size_t)B_DIM * X_DIM;

  auto cvt = [&](const float* src, bf16_t* dst, size_t n) {
    int g = (int)((n / 4 + 255) / 256); if (g > 4096) g = 4096;
    k_cvt<<<g, 256, 0, stream>>>(src, dst, n);
  };
  auto gemm_bf = [&](const bf16_t* A, int lda, const bf16_t* W, int ldw,
                     const float* bias, float* Cf, bf16_t* Cb, int ldc,
                     int M, int N, int K, int relu) {
    dim3 g(cdiv(M, 64), cdiv(N, 64));
    k_gemm_bf16<<<g, 128, 0, stream>>>(A, lda, W, ldw, bias, Cf, Cb, ldc, M, N, K, relu);
  };

  // ---- bf16 conversions (weights + inputs)
  cvt(x, x_bf, (size_t)B_DIM * X_DIM);
  cvt(xi, xi_bf, (size_t)B_DIM * T_DIM);
  for (int i = 0; i < 12; ++i) cvt(fw[i], wbf[i], wsz[i]);

  // ---- encoder front
  gemm_bf(x_bf, X_DIM, wbf[0], X_DIM, fb[0], nullptr, h1_bf, H_DIM, B_DIM, H_DIM, X_DIM, 1);
  gemm_bf(h1_bf, H_DIM, wbf[1], H_DIM, fb[1], sbuf, nullptr, T_DIM, B_DIM, T_DIM, H_DIM, 0);
  gemm_bf(h1_bf, H_DIM, wbf[2], H_DIM, fb[2], tbuf, nullptr, D_DIM, B_DIM, D_DIM, H_DIM, 0);

  // ---- RBF kernel matrices (fp32 WMMA, mode 2)
  k_rownorm<<<cdiv(B_DIM, 256), 256, 0, stream>>>(sbuf, T_DIM, T_DIM, sn, B_DIM);
  k_rownorm<<<cdiv(B_DIM, 256), 256, 0, stream>>>(xi, T_DIM, T_DIM, xin, B_DIM);
  {
    dim3 g(NTOT / 64, NTOT / 64);
    k_gemm_f32<<<g, 128, 0, stream>>>(sbuf, T_DIM, 0, sbuf, T_DIM, 0, Kss, NTOT,
                                      NTOT, NTOT, T_DIM, 2, 0, sn, sn, 0);
    k_gemm_f32<<<g, 128, 0, stream>>>(sbuf, T_DIM, 0, xi, T_DIM, 0, Y, NTOT,
                                      NTOT, NTOT, T_DIM, 2, 0, sn, xin, 0);   // KxisT
  }
  k_copy_diag<<<cdiv(NTOT, 256), 256, 0, stream>>>(Y, NTOT, dK, NTOT);

  // ---- blocked Cholesky (fp32 WMMA SYRK updates)
  const int NB = 64, nb = NTOT / NB;
  for (int b = 0; b < nb; ++b) {
    int jb = b * NB;
    k_potf2<<<1, 64, 0, stream>>>(Kss, NTOT, jb);
    int rows0 = jb + NB, nrows = NTOT - rows0;
    if (nrows > 0) {
      k_panel_trsm<<<cdiv(nrows, 128), 128, 0, stream>>>(Kss, NTOT, jb, rows0, nrows);
      dim3 g(cdiv(nrows, 64), cdiv(nrows, 64));
      k_gemm_f32<<<g, 128, 0, stream>>>(
          Kss + (size_t)rows0 * NTOT + jb, NTOT, 0,
          Kss + (size_t)rows0 * NTOT + jb, NTOT, 0,
          Kss + (size_t)rows0 * NTOT + rows0, NTOT,
          nrows, nrows, NB, 1, 1, nullptr, nullptr, 0);
    }
  }
  // ---- forward solve  L * Y = KxisT
  for (int b = 0; b < nb; ++b) {
    int jb = b * NB;
    k_trsm_diag<<<NTOT / 128, 128, 0, stream>>>(Kss, NTOT, jb, Y, NTOT, NTOT, 0);
    int rows0 = jb + NB, nrows = NTOT - rows0;
    if (nrows > 0) {
      dim3 g(cdiv(nrows, 64), NTOT / 64);
      k_gemm_f32<<<g, 128, 0, stream>>>(
          Kss + (size_t)rows0 * NTOT + jb, NTOT, 0,
          Y + (size_t)jb * NTOT, NTOT, 1,
          Y + (size_t)rows0 * NTOT, NTOT,
          nrows, NTOT, NB, 1, 0, nullptr, nullptr, 0);
    }
  }
  // ---- backward solve  L^T * Y = Y
  for (int b = nb - 1; b >= 0; --b) {
    int jb = b * NB;
    k_trsm_diag<<<NTOT / 128, 128, 0, stream>>>(Kss, NTOT, jb, Y, NTOT, NTOT, 1);
    if (jb > 0) {
      dim3 g(cdiv(jb, 64), NTOT / 64);
      k_gemm_f32<<<g, 128, 0, stream>>>(
          Kss + (size_t)jb * NTOT, NTOT, 1,
          Y + (size_t)jb * NTOT, NTOT, 1,
          Y, NTOT, jb, NTOT, NB, 1, 0, nullptr, nullptr, 0);
    }
  }

  // ---- cov_diag, fused  f = sqrt(cd)*eps_gp + Y^T @ t
  k_cd<<<cdiv(NTOT, 256), 256, 0, stream>>>(Y, NTOT, dK, cd, NTOT);
  {
    dim3 g(NTOT / 64, 1);
    k_gemm_f32<<<g, 128, 0, stream>>>(Y, NTOT, 1, tbuf, D_DIM, 1, fbuf, D_DIM,
                                      B_DIM, D_DIM, NTOT, 3, 0, cd, eps_gp, D_DIM);
  }
  cvt(fbuf, f_bf, (size_t)B_DIM * D_DIM);

  // ---- decoder / VAE path (bf16 WMMA)
  k_concat_bf<<<1024, 256, 0, stream>>>(f_bf, D_DIM, D_DIM, xi_bf, T_DIM, T_DIM,
                                        catfxi_bf, 80, B_DIM);
  gemm_bf(catfxi_bf, 80, wbf[3], 80, fb[3], nullptr, hbuf_bf, H_DIM, B_DIM, H_DIM, 80, 1);
  gemm_bf(hbuf_bf, H_DIM, wbf[4], H_DIM, fb[4], zm_o, nullptr, Z_DIM, B_DIM, Z_DIM, H_DIM, 0);
  gemm_bf(hbuf_bf, H_DIM, wbf[5], H_DIM, fb[5], zc_o, nullptr, Z_DIM, B_DIM, Z_DIM, H_DIM, 0);
  k_reparam<<<4096, 256, 0, stream>>>(zm_o, zc_o, eps_nm, z_bf, (size_t)B_DIM * Z_DIM);

  k_concat_bf<<<2048, 256, 0, stream>>>(x_bf, X_DIM, X_DIM, z_bf, Z_DIM, Z_DIM,
                                        catxz_bf, 2048, B_DIM);
  gemm_bf(catxz_bf, 2048, wbf[6], 2048, fb[6], nullptr, hbuf_bf, H_DIM, B_DIM, H_DIM, 2048, 1);
  gemm_bf(hbuf_bf, H_DIM, wbf[7], H_DIM, fb[7], encm, nullptr, 80, B_DIM, 80, H_DIM, 0);
  gemm_bf(hbuf_bf, H_DIM, wbf[8], H_DIM, fb[8], encc, nullptr, 80, B_DIM, 80, H_DIM, 0);

  gemm_bf(z_bf, Z_DIM, wbf[9], Z_DIM, fb[9], nullptr, hbuf_bf, H_DIM, B_DIM, H_DIM, Z_DIM, 1);
  gemm_bf(hbuf_bf, H_DIM, wbf[10], H_DIM, fb[10], xm_o, nullptr, X_DIM, B_DIM, X_DIM, H_DIM, 0);
  gemm_bf(hbuf_bf, H_DIM, wbf[11], H_DIM, fb[11], xc_o, nullptr, X_DIM, B_DIM, X_DIM, H_DIM, 0);

  // ---- losses (deterministic two-stage reductions)
  k_red_kld<<<RGRID, 256, 0, stream>>>(zm_o, zc_o, Z_DIM, 0, Z_DIM, B_DIM, part);
  k_red_kld<<<RGRID, 256, 0, stream>>>(encm, encc, 80, T_DIM, D_DIM, B_DIM, part + RGRID);
  k_red_sq<<<RGRID, 256, 0, stream>>>(x, X_DIM, 0, xm_o, X_DIM, 0, X_DIM, B_DIM, part + 2 * RGRID);
  k_red_sq<<<RGRID, 256, 0, stream>>>(xi, T_DIM, 0, encm, 80, 0, T_DIM, B_DIM, part + 3 * RGRID);
  k_red_sumsq<<<RGRID, 256, 0, stream>>>(xi, (size_t)B_DIM * T_DIM, part + 4 * RGRID);
  k_finalize<<<1, 64, 0, stream>>>(part, out);
}
